// Decoder_17489106830107
// MI455X (gfx1250) — compile-verified, matrix operand order checked by reading, not run on hardware
//
#include <hip/hip_runtime.h>

// ---- problem dims ----
#define B_   2
#define C_   256
#define H_   64
#define W_   64
#define D_   64
#define DI_  128
#define N_   16
#define R_   4
#define K_   4
#define L_   4096      // H*W
#define P_   8192      // B*L
#define C3_  768       // 3*C
#define DM_  256       // 4*D
#define CP_  48        // padded R+2N (36 -> 48)

typedef __attribute__((ext_vector_type(16))) __bf16 v16bf;
typedef __attribute__((ext_vector_type(8)))  float  v8f;

#if __has_builtin(__builtin_amdgcn_global_load_async_to_lds_b128) && \
    __has_builtin(__builtin_amdgcn_s_wait_asynccnt)
#define HAVE_ASYNC_LDS 1
typedef int v4i_ __attribute__((vector_size(16)));
typedef __attribute__((address_space(1))) v4i_ gv4i;   // global (__device__) v4i
typedef __attribute__((address_space(3))) v4i_ lv4i;   // LDS (__shared__) v4i
#else
#define HAVE_ASYNC_LDS 0
#endif

__device__ __forceinline__ __bf16 f2bf(float f) {
  union { float f; unsigned u; } a; a.f = f;
  unsigned r = a.u + 0x7FFFu + ((a.u >> 16) & 1u);
  unsigned short hs = (unsigned short)(r >> 16);
  __bf16 o; __builtin_memcpy(&o, &hs, 2); return o;
}

// 16-bit A/B fragment K-index for element i (0..15) of v16bf, per ISA layout:
// lane<16: K in {0..7, 16..23}; lane>=16: K in {8..15, 24..31}
__device__ __forceinline__ int fkidx(int lane, int i) {
  return i + ((i & 8) ? 8 : 0) + ((lane & 16) ? 8 : 0);
}

__device__ __forceinline__ v8f vzero8() {
  v8f z;
#pragma unroll
  for (int r = 0; r < 8; ++r) z[r] = 0.f;
  return z;
}

__device__ __forceinline__ float silu_f(float x) {
  return x / (1.f + __expf(-x));
}

// ---------------- K0: weight conversion ----------------
__global__ __launch_bounds__(256) void k0_cvt(const float* __restrict__ src,
                                              __bf16* __restrict__ dst, int n) {
  int i = blockIdx.x * 256 + threadIdx.x;
  if (i < n) dst[i] = f2bf(src[i]);
}

__global__ __launch_bounds__(256) void k0_cvt_xproj(const float* __restrict__ src,
                                                    __bf16* __restrict__ dst) {
  // src [K][36][DI], dst [K][48][DI] zero-padded
  int i = blockIdx.x * 256 + threadIdx.x;
  if (i >= K_ * CP_ * DI_) return;
  int d = i % DI_; int rest = i / DI_;
  int c = rest % CP_; int k = rest / CP_;
  float v = (c < (R_ + 2 * N_)) ? src[(k * (R_ + 2 * N_) + c) * DI_ + d] : 0.f;
  dst[i] = f2bf(v);
}

// ---------------- K1: conv1x1 GEMM + LN1 + in_proj GEMM + silu(z) ----------------
__global__ __launch_bounds__(128)
void k1_conv_ln_inproj(const float* __restrict__ x1, const float* __restrict__ x2,
                       const float* __restrict__ x3,
                       const __bf16* __restrict__ convw_bf,   // [64][768]
                       const float* __restrict__ conv_b,
                       const float* __restrict__ ln1_g, const float* __restrict__ ln1_b,
                       const __bf16* __restrict__ inproj_bf,  // [64][256]
                       float* __restrict__ x_res,             // [P][64]
                       float* __restrict__ xp,                // [B][DI][L]
                       float* __restrict__ silu_z)            // [P][DI]
{
  __shared__ __bf16 Asub[16][C3_ + 8];   // padded: row stride skews banks
  __shared__ float  xld[16][D_ + 4];
  __shared__ __bf16 xnb[16][D_ + 8];

  int tid = threadIdx.x;
  int lane = tid & 31, wv = tid >> 5;
  int pix0 = blockIdx.x * 16;
  int bi = pix0 / L_;
  int hw0 = pix0 % L_;

  // cooperative A load: 16 px x 768 ch, coalesced over hw
  for (int idx = tid; idx < 16 * C3_; idx += 128) {
    int p = idx & 15, c = idx >> 4;
    const float* src = (c < C_) ? x1 : ((c < 2 * C_) ? x2 : x3);
    int cm = c & (C_ - 1);
    Asub[p][c] = f2bf(src[(bi * C_ + cm) * L_ + hw0 + p]);
  }
  __syncthreads();

  int row = lane & 15;
  int nn = lane & 15, half = lane >> 4;

  // GEMM1: [16 x 768] x [768 x 64]; wave wv -> out cols [16wv, 16wv+16)
  {
    v8f acc = vzero8();
    int colo = wv * 16 + nn;
    for (int kb = 0; kb < C3_; kb += 32) {
      v16bf a, bm;
#pragma unroll
      for (int i = 0; i < 16; ++i) {
        int kk = fkidx(lane, i);
        a[i]  = Asub[row][kb + kk];
        bm[i] = convw_bf[colo * C3_ + kb + kk];
      }
      acc = __builtin_amdgcn_wmma_f32_16x16x32_bf16(false, a, false, bm,
                                                    (short)0, acc, false, false);
    }
#pragma unroll
    for (int r = 0; r < 8; ++r) {
      int m = r + half * 8;
      xld[m][wv * 16 + nn] = acc[r] + conv_b[wv * 16 + nn];
    }
  }
  __syncthreads();

  // LN1 + residual store
  if (tid < 16) {
    int m = tid;
    float mu = 0.f;
    for (int j = 0; j < D_; ++j) mu += xld[m][j];
    mu *= (1.f / D_);
    float var = 0.f;
    for (int j = 0; j < D_; ++j) { float d = xld[m][j] - mu; var += d * d; }
    var *= (1.f / D_);
    float rs = rsqrtf(var + 1e-5f);
    for (int j = 0; j < D_; ++j) {
      float xv = xld[m][j];
      x_res[(pix0 + m) * D_ + j] = xv;
      xnb[m][j] = f2bf((xv - mu) * rs * ln1_g[j] + ln1_b[j]);
    }
  }
  __syncthreads();

  // GEMM2: [16 x 64] x [64 x 256]; wave wv -> cols [64wv, 64wv+64)
  for (int j = 0; j < 4; ++j) {
    int cb = wv * 64 + j * 16;
    v8f acc = vzero8();
    for (int kb = 0; kb < D_; kb += 32) {
      v16bf a, bm;
#pragma unroll
      for (int i = 0; i < 16; ++i) {
        int kk = fkidx(lane, i);
        a[i]  = xnb[row][kb + kk];
        bm[i] = inproj_bf[(kb + kk) * (2 * DI_) + cb + nn];
      }
      acc = __builtin_amdgcn_wmma_f32_16x16x32_bf16(false, a, false, bm,
                                                    (short)0, acc, false, false);
    }
    int col = cb + nn;
#pragma unroll
    for (int r = 0; r < 8; ++r) {
      int m = r + half * 8;
      float v = acc[r];
      if (col < DI_)
        xp[((size_t)bi * DI_ + col) * L_ + hw0 + m] = v;
      else
        silu_z[(size_t)(pix0 + m) * DI_ + (col - DI_)] = silu_f(v);
    }
  }
}

// ---------------- K2: depthwise 3x3 + bias + SiLU ----------------
__global__ __launch_bounds__(256)
void k2_dwconv(const float* __restrict__ xp, const float* __restrict__ dww,
               const float* __restrict__ dwb,
               float* __restrict__ xc, __bf16* __restrict__ xcbf) {
  int idx = blockIdx.x * 256 + threadIdx.x;
  if (idx >= B_ * DI_ * L_) return;
  int l = idx & (L_ - 1); int bd = idx / L_;
  int d = bd % DI_; int bi = bd / DI_;
  int h = l >> 6, w = l & 63;
  float s = dwb[d];
  const float* base = xp + ((size_t)bi * DI_ + d) * L_;
#pragma unroll
  for (int dh = -1; dh <= 1; ++dh) {
    int h2 = h + dh;
    if (h2 < 0 || h2 >= H_) continue;
#pragma unroll
    for (int dw = -1; dw <= 1; ++dw) {
      int w2 = w + dw;
      if (w2 < 0 || w2 >= W_) continue;
      s += base[h2 * W_ + w2] * dww[d * 9 + (dh + 1) * 3 + (dw + 1)];
    }
  }
  float v = silu_f(s);
  xc[idx] = v;
  xcbf[idx] = f2bf(v);
}

// ---------------- K3: x_proj GEMM per (b,k): [L x DI] x [DI x 48] ----------------
__global__ __launch_bounds__(128)
void k3_xproj(const __bf16* __restrict__ xcbf,      // [B][DI][L]
              const __bf16* __restrict__ xprojbf,   // [K][48][DI]
              float* __restrict__ x_dbl)            // [B][K][48][L]
{
  __shared__ __bf16 xt[DI_][64];   // 16 KB: [d][l - l0]
  int tid = threadIdx.x, lane = tid & 31, wv = tid >> 5;
  int blk = blockIdx.x;
  int lt = blk % (L_ / 64); int bk = blk / (L_ / 64);
  int k = bk % K_; int bi = bk / K_;
  int l0 = lt * 64;

#if HAVE_ASYNC_LDS
  // 16 KB global->LDS copy with no conversion: use async-to-LDS (ASYNCcnt path),
  // 128 threads x 8 iters x 16B, per-lane contiguous b128 transfers.
  for (int idx = tid; idx < DI_ * 8; idx += 128) {
    int grp = idx & 7, d = idx >> 3;
    __bf16* ldst = &xt[d][grp * 8];
    const __bf16* gsrc = &xcbf[((size_t)bi * DI_ + d) * L_ + l0 + grp * 8];
    __builtin_amdgcn_global_load_async_to_lds_b128(
        (gv4i*)(void*)gsrc, (lv4i*)(void*)ldst, 0, 0);
  }
  __builtin_amdgcn_s_wait_asynccnt(0);
  __syncthreads();
#else
  for (int idx = tid; idx < DI_ * 64; idx += 128) {
    int li = idx & 63, d = idx >> 6;
    xt[d][li] = xcbf[((size_t)bi * DI_ + d) * L_ + l0 + li];
  }
  __syncthreads();
#endif

  int row = lane & 15, nn = lane & 15, half = lane >> 4;
  for (int ct = 0; ct < 3; ++ct) {
    v8f acc = vzero8();
    for (int kb = 0; kb < DI_; kb += 32) {
      v16bf a, bm;
#pragma unroll
      for (int i = 0; i < 16; ++i) {
        int kk = fkidx(lane, i);
        a[i]  = xt[kb + kk][wv * 16 + row];
        bm[i] = xprojbf[(k * CP_ + ct * 16 + nn) * DI_ + kb + kk];
      }
      acc = __builtin_amdgcn_wmma_f32_16x16x32_bf16(false, a, false, bm,
                                                    (short)0, acc, false, false);
    }
#pragma unroll
    for (int r = 0; r < 8; ++r) {
      int m = r + half * 8;
      int c = ct * 16 + nn;
      x_dbl[(((size_t)bi * K_ + k) * CP_ + c) * L_ + l0 + wv * 16 + m] = acc[r];
    }
  }
}

// ---------------- K4: selective scan ----------------
// one wave per (b,k,d-pair); lane n<16 owns state n of d0, lanes 16..31 of d1
__global__ __launch_bounds__(32)
void k4_scan(const float* __restrict__ x_dbl,  // [B][K][48][L] (xc-order)
             const float* __restrict__ xc,     // [B][DI][L]
             const float* __restrict__ A_logs, // [K*DI][N]
             const float* __restrict__ Ds,     // [K*DI]
             const float* __restrict__ dtw,    // [K][DI][R]
             const float* __restrict__ dtb,    // [K][DI]
             float* __restrict__ ys)           // [B][K][DI][L] (scan-order)
{
  int lane = threadIdx.x;
  int n = lane & 15, dsel = lane >> 4;
  int blk = blockIdx.x;                       // B*K*(DI/2)
  int dp = blk % (DI_ / 2); int bk = blk / (DI_ / 2);
  int k = bk % K_; int bi = bk / K_;
  int d = dp * 2 + dsel;
  int kd = k * DI_ + d;

  float Aval = -__expf(A_logs[kd * N_ + n]);
  float Dsk  = Ds[kd];
  float db   = dtb[kd];
  float w0 = dtw[kd * R_ + 0], w1 = dtw[kd * R_ + 1];
  float w2 = dtw[kd * R_ + 2], w3 = dtw[kd * R_ + 3];

  const float* xd = x_dbl + ((size_t)bi * K_ + k) * CP_ * L_;
  const float* uc = xc + ((size_t)bi * DI_ + d) * L_;
  float* yo = ys + (((size_t)bi * K_ + k) * DI_ + d) * L_;

  const bool trans = (k & 1);
  const bool flip  = (k & 2);

  float h = 0.f;
  for (int l = 0; l < L_; ++l) {
    int ls  = flip ? (L_ - 1 - l) : l;
    int src = trans ? (((ls & 63) << 6) | (ls >> 6)) : ls;
    if ((l & 31) == 0) __builtin_prefetch(uc + src, 0, 1);

    float u = uc[src];
    float draw = w0 * xd[0 * L_ + src] + w1 * xd[1 * L_ + src]
               + w2 * xd[2 * L_ + src] + w3 * xd[3 * L_ + src] + db;
    float delta = (draw > 20.f) ? draw : log1pf(__expf(draw));
    float Bv = xd[(R_ + n) * L_ + src];
    float Cv = xd[(R_ + N_ + n) * L_ + src];
    float dA = __expf(delta * Aval);
    h = h * dA + (delta * u) * Bv;
    float yp = h * Cv;
    yp += __shfl_xor(yp, 1, 16);
    yp += __shfl_xor(yp, 2, 16);
    yp += __shfl_xor(yp, 4, 16);
    yp += __shfl_xor(yp, 8, 16);
    if (n == 0) yo[l] = yp + u * Dsk;
  }
}

// ---------------- K5: merge + out_norm + out_proj + MLP + residual ----------------
__global__ __launch_bounds__(128)
void k5_tail(const float* __restrict__ ys,       // [B][K][DI][L]
             const float* __restrict__ silu_z,   // [P][DI]
             const float* __restrict__ x_res,    // [P][64]
             const float* __restrict__ onorm_g, const float* __restrict__ onorm_b,
             const __bf16* __restrict__ outproj_bf,  // [128][64]
             const float* __restrict__ ln2_g, const float* __restrict__ ln2_b,
             const __bf16* __restrict__ fc1_bf,      // [64][256]
             const float* __restrict__ fc1_b,
             const __bf16* __restrict__ fc2_bf,      // [256][64]
             const float* __restrict__ fc2_b,
             float* __restrict__ out)                // [B][64][L]
{
  __shared__ float  yln[16][DI_ + 4];
  __shared__ __bf16 ybf[16][DI_ + 8];
  __shared__ float  x2[16][D_ + 4];
  __shared__ __bf16 hbf[16][D_ + 8];
  __shared__ __bf16 gbf[16][DM_ + 8];

  int tid = threadIdx.x, lane = tid & 31, wv = tid >> 5;
  int pix0 = blockIdx.x * 16;
  int bi = pix0 / L_, hw0 = pix0 % L_;
  int row = lane & 15, nn = lane & 15, half = lane >> 4;

  // cross-merge gather (all direction maps are involutions)
  for (int idx = tid; idx < 16 * DI_; idx += 128) {
    int m = idx & 15, d = idx >> 4;
    int pos = hw0 + m;
    int pt = ((pos & 63) << 6) | (pos >> 6);
    size_t base = ((size_t)bi * K_) * DI_ * L_;
    float v = ys[base + ((size_t)0 * DI_ + d) * L_ + pos]
            + ys[base + ((size_t)1 * DI_ + d) * L_ + pt]
            + ys[base + ((size_t)2 * DI_ + d) * L_ + (L_ - 1 - pos)]
            + ys[base + ((size_t)3 * DI_ + d) * L_ + (L_ - 1 - pt)];
    yln[m][d] = v;
  }
  __syncthreads();

  // out_norm + gate
  if (tid < 16) {
    int m = tid;
    float mu = 0.f;
    for (int j = 0; j < DI_; ++j) mu += yln[m][j];
    mu *= (1.f / DI_);
    float var = 0.f;
    for (int j = 0; j < DI_; ++j) { float d = yln[m][j] - mu; var += d * d; }
    var *= (1.f / DI_);
    float rs = rsqrtf(var + 1e-5f);
    for (int j = 0; j < DI_; ++j) {
      float t = (yln[m][j] - mu) * rs * onorm_g[j] + onorm_b[j];
      t *= silu_z[(size_t)(pix0 + m) * DI_ + j];
      ybf[m][j] = f2bf(t);
    }
  }
  __syncthreads();

  // out_proj GEMM: [16 x 128] x [128 x 64]; wave wv -> col tile wv
  {
    v8f acc = vzero8();
    for (int kb = 0; kb < DI_; kb += 32) {
      v16bf a, bm;
#pragma unroll
      for (int i = 0; i < 16; ++i) {
        int kk = fkidx(lane, i);
        a[i]  = ybf[row][kb + kk];
        bm[i] = outproj_bf[(kb + kk) * D_ + wv * 16 + nn];
      }
      acc = __builtin_amdgcn_wmma_f32_16x16x32_bf16(false, a, false, bm,
                                                    (short)0, acc, false, false);
    }
#pragma unroll
    for (int r = 0; r < 8; ++r) {
      int m = r + half * 8;
      x2[m][wv * 16 + nn] = acc[r] + x_res[(size_t)(pix0 + m) * D_ + wv * 16 + nn];
    }
  }
  __syncthreads();

  // LN2
  if (tid < 16) {
    int m = tid;
    float mu = 0.f;
    for (int j = 0; j < D_; ++j) mu += x2[m][j];
    mu *= (1.f / D_);
    float var = 0.f;
    for (int j = 0; j < D_; ++j) { float d = x2[m][j] - mu; var += d * d; }
    var *= (1.f / D_);
    float rs = rsqrtf(var + 1e-5f);
    for (int j = 0; j < D_; ++j)
      hbf[m][j] = f2bf((x2[m][j] - mu) * rs * ln2_g[j] + ln2_b[j]);
  }
  __syncthreads();

  // fc1 GEMM: [16 x 64] x [64 x 256] + bias + gelu(tanh)
  for (int j = 0; j < 4; ++j) {
    int cb = wv * 64 + j * 16;
    v8f acc = vzero8();
    for (int kb = 0; kb < D_; kb += 32) {
      v16bf a, bm;
#pragma unroll
      for (int i = 0; i < 16; ++i) {
        int kk = fkidx(lane, i);
        a[i]  = hbf[row][kb + kk];
        bm[i] = fc1_bf[(kb + kk) * DM_ + cb + nn];
      }
      acc = __builtin_amdgcn_wmma_f32_16x16x32_bf16(false, a, false, bm,
                                                    (short)0, acc, false, false);
    }
#pragma unroll
    for (int r = 0; r < 8; ++r) {
      int m = r + half * 8;
      float v = acc[r] + fc1_b[cb + nn];
      float g = 0.5f * v * (1.f + tanhf(0.7978845608f * (v + 0.044715f * v * v * v)));
      gbf[m][cb + nn] = f2bf(g);
    }
  }
  __syncthreads();

  // fc2 GEMM: [16 x 256] x [256 x 64] + bias + residual; store transposed out
  {
    v8f acc = vzero8();
    for (int kb = 0; kb < DM_; kb += 32) {
      v16bf a, bm;
#pragma unroll
      for (int i = 0; i < 16; ++i) {
        int kk = fkidx(lane, i);
        a[i]  = gbf[row][kb + kk];
        bm[i] = fc2_bf[(kb + kk) * D_ + wv * 16 + nn];
      }
      acc = __builtin_amdgcn_wmma_f32_16x16x32_bf16(false, a, false, bm,
                                                    (short)0, acc, false, false);
    }
    int col = wv * 16 + nn;
#pragma unroll
    for (int r = 0; r < 8; ++r) {
      int m = r + half * 8;
      float v = acc[r] + fc2_b[col] + x2[m][col];
      out[((size_t)bi * D_ + col) * L_ + hw0 + m] = v;
    }
  }
}

// ---------------- host launcher ----------------
extern "C" void kernel_launch(void* const* d_in, const int* in_sizes, int n_in,
                              void* d_out, int out_size, void* d_ws, size_t ws_size,
                              hipStream_t stream) {
  const float* x1        = (const float*)d_in[0];
  const float* x2        = (const float*)d_in[1];
  const float* x3        = (const float*)d_in[2];
  const float* conv_w    = (const float*)d_in[3];
  const float* conv_b    = (const float*)d_in[4];
  const float* ln1_g     = (const float*)d_in[5];
  const float* ln1_b     = (const float*)d_in[6];
  const float* in_proj_w = (const float*)d_in[7];
  const float* dw_w      = (const float*)d_in[8];
  const float* dw_b      = (const float*)d_in[9];
  const float* x_proj_w  = (const float*)d_in[10];
  const float* dt_proj_w = (const float*)d_in[11];
  const float* dt_proj_b = (const float*)d_in[12];
  const float* A_logs    = (const float*)d_in[13];
  const float* Ds        = (const float*)d_in[14];
  const float* onorm_g   = (const float*)d_in[15];
  const float* onorm_b   = (const float*)d_in[16];
  const float* out_proj_w= (const float*)d_in[17];
  const float* ln2_g     = (const float*)d_in[18];
  const float* ln2_b     = (const float*)d_in[19];
  const float* fc1_w     = (const float*)d_in[20];
  const float* fc1_b     = (const float*)d_in[21];
  const float* fc2_w     = (const float*)d_in[22];
  const float* fc2_b     = (const float*)d_in[23];

  char* ws = (char*)d_ws;
  size_t off = 0;
  auto alloc = [&](size_t bytes) -> char* {
    char* p = ws + off;
    off += (bytes + 255) & ~(size_t)255;
    return p;
  };

  __bf16* convw_bf   = (__bf16*)alloc((size_t)D_ * C3_ * 2);
  __bf16* inproj_bf  = (__bf16*)alloc((size_t)D_ * 2 * DI_ * 2);
  __bf16* xproj_bf   = (__bf16*)alloc((size_t)K_ * CP_ * DI_ * 2);
  __bf16* outproj_bf = (__bf16*)alloc((size_t)DI_ * D_ * 2);
  __bf16* fc1_bf     = (__bf16*)alloc((size_t)D_ * DM_ * 2);
  __bf16* fc2_bf     = (__bf16*)alloc((size_t)DM_ * D_ * 2);
  float*  x_res      = (float*)alloc((size_t)P_ * D_ * 4);
  float*  silu_z     = (float*)alloc((size_t)P_ * DI_ * 4);
  float*  xp         = (float*)alloc((size_t)B_ * DI_ * L_ * 4);
  float*  xc         = (float*)alloc((size_t)B_ * DI_ * L_ * 4);
  __bf16* xcbf       = (__bf16*)alloc((size_t)B_ * DI_ * L_ * 2);
  float*  x_dbl      = (float*)alloc((size_t)B_ * K_ * CP_ * L_ * 4);
  float*  ys         = (float*)alloc((size_t)B_ * K_ * DI_ * L_ * 4);
  (void)ws_size; (void)in_sizes; (void)n_in; (void)out_size;

  // K0: weight conversion
  k0_cvt<<<(D_ * C3_ + 255) / 256, 256, 0, stream>>>(conv_w, convw_bf, D_ * C3_);
  k0_cvt<<<(D_ * 2 * DI_ + 255) / 256, 256, 0, stream>>>(in_proj_w, inproj_bf, D_ * 2 * DI_);
  k0_cvt_xproj<<<(K_ * CP_ * DI_ + 255) / 256, 256, 0, stream>>>(x_proj_w, xproj_bf);
  k0_cvt<<<(DI_ * D_ + 255) / 256, 256, 0, stream>>>(out_proj_w, outproj_bf, DI_ * D_);
  k0_cvt<<<(D_ * DM_ + 255) / 256, 256, 0, stream>>>(fc1_w, fc1_bf, D_ * DM_);
  k0_cvt<<<(DM_ * D_ + 255) / 256, 256, 0, stream>>>(fc2_w, fc2_bf, DM_ * D_);

  // K1: conv1x1 + LN1 + in_proj
  k1_conv_ln_inproj<<<P_ / 16, 128, 0, stream>>>(x1, x2, x3, convw_bf, conv_b,
                                                 ln1_g, ln1_b, inproj_bf,
                                                 x_res, xp, silu_z);
  // K2: depthwise conv + SiLU
  k2_dwconv<<<(B_ * DI_ * L_) / 256, 256, 0, stream>>>(xp, dw_w, dw_b, xc, xcbf);
  // K3: x_proj GEMM
  k3_xproj<<<B_ * K_ * (L_ / 64), 128, 0, stream>>>(xcbf, xproj_bf, x_dbl);
  // K4: selective scan
  k4_scan<<<B_ * K_ * (DI_ / 2), 32, 0, stream>>>(x_dbl, xc, A_logs, Ds,
                                                  dt_proj_w, dt_proj_b, ys);
  // K5: merge + out_norm + out_proj + MLP
  k5_tail<<<P_ / 16, 128, 0, stream>>>(ys, silu_z, x_res, onorm_g, onorm_b,
                                       outproj_bf, ln2_g, ln2_b,
                                       fc1_bf, fc1_b, fc2_bf, fc2_b,
                                       (float*)d_out);
}
